// GNN_60318520705103
// MI455X (gfx1250) — matrix operand order repaired and checked by reading, not compiled, not run on hardware
//
#include <hip/hip_runtime.h>
#include <hip/hip_bf16.h>

// ---------------- problem constants (fixed by the reference) ----------------
#define ND_    8192
#define NP__   8192
#define ED_    262144
#define EP__   262144
#define BATCH  4096
#define HID    512
#define DIMF   3372           // true feature width
#define DIMP   3392           // padded to multiple of 32 (K for WMMA)
#define O1     1024
#define O2     256
#define EPS_BN 1e-5f
#define EPB    8              // edges per block in scatter-add

typedef unsigned short u16;
typedef __attribute__((ext_vector_type(16))) __bf16 bf16x16;
typedef __attribute__((ext_vector_type(2)))  __bf16 bf16x2;
typedef __attribute__((ext_vector_type(8)))  float  f32x8;

// native bf16 converts -> v_cvt_pk_bf16_f32
__device__ __forceinline__ unsigned int pack2bf(float x, float y) {
  bf16x2 h; h[0] = (__bf16)x; h[1] = (__bf16)y;
  return __builtin_bit_cast(unsigned int, h);
}

// ============================================================================
// Tiled WMMA GEMM: C[M,N] = A[M,K] * B[K,N]  (A optionally a transposed view of
// a [K,M] row-major buffer). fp32 in global, bf16 in LDS (double buffered),
// fp32 accumulate via v_wmma_f32_16x16x32_bf16.
// Block tile 128x128, 8 waves (2x4), wave tile 64x32 (4x2 WMMA tiles).
// Two-deep software pipeline: regs(t+2) <- global while LDS(t+1) <- regs and
// WMMA consumes LDS(t).
// Requires: M%128==0, N%128==0, K%32==0 (guaranteed by caller).
// EPI: 0 = store, 1 = +bias, 2 = +bias then ReLU.
// ============================================================================
struct TileRegs { float4 a[4]; float4 b[4]; };

template<bool TRANS_A>
__device__ __forceinline__ void load_tile_regs(TileRegs& t,
                                               const float* __restrict__ A,
                                               const float* __restrict__ B,
                                               int m0, int n0, int k0,
                                               int lda, int ldb, int tid)
{
  if (!TRANS_A) {
    #pragma unroll
    for (int it = 0; it < 4; ++it) {
      int i = tid + it * 256;            // 0..1023
      int r = i >> 3;                    // 0..127
      int c = (i & 7) << 2;              // 0,4,...,28
      t.a[it] = *reinterpret_cast<const float4*>(
          A + (size_t)(m0 + r) * lda + (k0 + c));
    }
  } else {
    #pragma unroll
    for (int it = 0; it < 2; ++it) {
      int i  = tid + it * 256;           // 0..511
      int rp = i >> 5;                   // K row-pair 0..15
      int c  = (i & 31) << 2;            // M offset 0..124
      t.a[2*it + 0] = *reinterpret_cast<const float4*>(
          A + (size_t)(k0 + 2*rp + 0) * lda + (m0 + c));
      t.a[2*it + 1] = *reinterpret_cast<const float4*>(
          A + (size_t)(k0 + 2*rp + 1) * lda + (m0 + c));
    }
  }
  #pragma unroll
  for (int it = 0; it < 2; ++it) {
    int i  = tid + it * 256;
    int rp = i >> 5;
    int c  = (i & 31) << 2;              // N offset
    t.b[2*it + 0] = *reinterpret_cast<const float4*>(
        B + (size_t)(k0 + 2*rp + 0) * ldb + (n0 + c));
    t.b[2*it + 1] = *reinterpret_cast<const float4*>(
        B + (size_t)(k0 + 2*rp + 1) * ldb + (n0 + c));
  }
}

template<bool TRANS_A>
__device__ __forceinline__ void store_tile_lds(const TileRegs& t,
                                               u16 (*sAb)[32], u16 (*sBb)[32],
                                               int tid)
{
  if (!TRANS_A) {
    #pragma unroll
    for (int it = 0; it < 4; ++it) {
      int i = tid + it * 256;
      int r = i >> 3;
      int c = (i & 7) << 2;
      uint2 p; p.x = pack2bf(t.a[it].x, t.a[it].y);
      p.y = pack2bf(t.a[it].z, t.a[it].w);
      *reinterpret_cast<uint2*>(&sAb[r][c]) = p;              // ds_store_b64
    }
  } else {
    #pragma unroll
    for (int it = 0; it < 2; ++it) {
      int i  = tid + it * 256;
      int rp = i >> 5;
      int c  = (i & 31) << 2;
      const float4 v0 = t.a[2*it + 0], v1 = t.a[2*it + 1];
      *reinterpret_cast<unsigned int*>(&sAb[c + 0][2*rp]) = pack2bf(v0.x, v1.x);
      *reinterpret_cast<unsigned int*>(&sAb[c + 1][2*rp]) = pack2bf(v0.y, v1.y);
      *reinterpret_cast<unsigned int*>(&sAb[c + 2][2*rp]) = pack2bf(v0.z, v1.z);
      *reinterpret_cast<unsigned int*>(&sAb[c + 3][2*rp]) = pack2bf(v0.w, v1.w);
    }
  }
  #pragma unroll
  for (int it = 0; it < 2; ++it) {
    int i  = tid + it * 256;
    int rp = i >> 5;
    int c  = (i & 31) << 2;
    const float4 v0 = t.b[2*it + 0], v1 = t.b[2*it + 1];
    *reinterpret_cast<unsigned int*>(&sBb[c + 0][2*rp]) = pack2bf(v0.x, v1.x);
    *reinterpret_cast<unsigned int*>(&sBb[c + 1][2*rp]) = pack2bf(v0.y, v1.y);
    *reinterpret_cast<unsigned int*>(&sBb[c + 2][2*rp]) = pack2bf(v0.z, v1.z);
    *reinterpret_cast<unsigned int*>(&sBb[c + 3][2*rp]) = pack2bf(v0.w, v1.w);
  }
}

template<bool TRANS_A, int EPI>
__global__ __launch_bounds__(256)
void wmma_gemm(const float* __restrict__ A, const float* __restrict__ B,
               const float* __restrict__ bias, float* __restrict__ C,
               int K, int lda, int ldb, int ldc)
{
  __shared__ alignas(16) u16 sA[2][128][32];   // A tiles, row-major [m][k]
  __shared__ alignas(16) u16 sB[2][128][32];   // B tiles, transposed [n][k]

  const int tid  = threadIdx.x;
  const int lane = tid & 31;
  const int wave = tid >> 5;
  const int wm   = wave >> 2;            // 0..1 (M direction)
  const int wn   = wave & 3;             // 0..3 (N direction)
  const int m0   = blockIdx.y * 128;
  const int n0   = blockIdx.x * 128;

  f32x8 acc[4][2];
  #pragma unroll
  for (int i = 0; i < 4; ++i)
    #pragma unroll
    for (int j = 0; j < 2; ++j)
      #pragma unroll
      for (int v = 0; v < 8; ++v) acc[i][j][v] = 0.f;

  const int frow = lane & 15;            // fragment row (M for A, N for B)
  const int kba  = (lane >> 4) << 3;     // A half-wave K base (0 / 8)
  const int kbb  = (lane >> 4) << 4;     // unused; B loads use kba (0 / 8)

  // ---- pipeline prologue ----
  const int ntiles = K >> 5;
  TileRegs tr;
  load_tile_regs<TRANS_A>(tr, A, B, m0, n0, 0, lda, ldb, tid);
  store_tile_lds<TRANS_A>(tr, sA[0], sB[0], tid);
  if (ntiles > 1)
    load_tile_regs<TRANS_A>(tr, A, B, m0, n0, 32, lda, ldb, tid);

  int cur = 0;
  for (int t = 0; t < ntiles; ++t, cur ^= 1) {
    __syncthreads();   // sm[cur] stores visible; sm[cur^1] fully consumed

    if (t + 1 < ntiles) {
      store_tile_lds<TRANS_A>(tr, sA[cur ^ 1], sB[cur ^ 1], tid);
      if (t + 2 < ntiles)
        load_tile_regs<TRANS_A>(tr, A, B, m0, n0, (t + 2) * 32, lda, ldb, tid);
      if (t + 3 < ntiles) {
        // L2 prefetch 3 tiles ahead (global_prefetch_b8)
        int kp = (t + 3) * 32;
        const float* pa = TRANS_A
            ? (A + (size_t)(kp + (tid >> 3)) * lda + m0 + ((tid & 7) << 4))
            : (A + (size_t)(m0 + (tid >> 1)) * lda + kp);
        __builtin_prefetch(pa, 0, 1);
        __builtin_prefetch(B + (size_t)(kp + (tid >> 3)) * ldb
                             + n0 + ((tid & 7) << 4), 0, 1);
      }
    }

    // ---- fragments (ISA 16-bit A 16x32 / B 32x16 layouts) ----
    union AF { bf16x16 v; u16 u[16]; };
    AF a[4], b[2];
    #pragma unroll
    for (int mt = 0; mt < 4; ++mt) {
      const u16* p = &sA[cur][wm * 64 + mt * 16 + frow][0];
      #pragma unroll
      for (int j = 0; j < 8; ++j) {
        a[mt].u[j]     = p[kba + j];
        a[mt].u[8 + j] = p[kba + 16 + j];
      }
    }
    #pragma unroll
    for (int nt = 0; nt < 2; ++nt) {
      const u16* p = &sB[cur][wn * 32 + nt * 16 + frow][0];
      #pragma unroll
      for (int j = 0; j < 8; ++j) {
        b[nt].u[j]     = p[kba + j];
        b[nt].u[8 + j] = p[kba + 16 + j];
      }
    }

    #pragma unroll
    for (int mt = 0; mt < 4; ++mt)
      #pragma unroll
      for (int nt = 0; nt < 2; ++nt)
        acc[mt][nt] = __builtin_amdgcn_wmma_f32_16x16x32_bf16(
            false, a[mt].v, false, b[nt].v, (short)0, acc[mt][nt], false, false);
  }

  // ---- epilogue: C layout M = 8*(lane>=16)+vgpr, N = lane&15 ----
  const int cr = (lane >> 4) << 3;
  const int cn = lane & 15;
  #pragma unroll
  for (int mt = 0; mt < 4; ++mt) {
    #pragma unroll
    for (int nt = 0; nt < 2; ++nt) {
      const int row = m0 + wm * 64 + mt * 16 + cr;
      const int col = n0 + wn * 32 + nt * 16 + cn;
      float bv = 0.f;
      if (EPI >= 1) bv = bias[col];
      #pragma unroll
      for (int v = 0; v < 8; ++v) {
        float x = acc[mt][nt][v] + bv;
        if (EPI == 2) x = fmaxf(x, 0.f);
        C[(size_t)(row + v) * ldc + col] = x;
      }
    }
  }
}

// ============================================================================
// GCN helper kernels
// ============================================================================
__global__ void k_fill1(float* p, int n) {
  int i = blockIdx.x * 256 + threadIdx.x;
  if (i < n) p[i] = 1.f;                         // self-loop weight
}

__global__ void k_deg(const int* __restrict__ ei, const float* __restrict__ ew,
                      float* deg, int E) {
  int e = blockIdx.x * 256 + threadIdx.x;
  if (e < E) atomicAdd(&deg[ei[E + e]], ew[e]);  // dst = ei[1][e]
}

__global__ void k_dinv(float* d, int n) {
  int i = blockIdx.x * 256 + threadIdx.x;
  if (i < n) { float v = d[i]; d[i] = (v > 0.f) ? rsqrtf(v) : 0.f; }
}

// agg init = self-loop contribution: h[i]*dinv[i]^2
__global__ void k_selfinit(const float* __restrict__ h, const float* __restrict__ dinv,
                           float* __restrict__ agg, int n) {
  int idx = blockIdx.x * 256 + threadIdx.x;
  if (idx < n) { int r = idx >> 9; float s = dinv[r]; agg[idx] = h[idx] * s * s; }
}

// block handles EPB edges x 256 columns: agg[dst] += h[src]*norm
__global__ void k_edgeagg(const int* __restrict__ ei, const float* __restrict__ ew,
                          const float* __restrict__ dinv, const float* __restrict__ h,
                          float* __restrict__ agg, int E) {
  int c  = blockIdx.y * 256 + threadIdx.x;
  int e0 = blockIdx.x * EPB;
  #pragma unroll
  for (int j = 0; j < EPB; ++j) {
    int e = e0 + j;
    int s = ei[e], d = ei[E + e];
    float nrm = dinv[s] * ew[e] * dinv[d];
    atomicAdd(&agg[(size_t)d * HID + c], h[(size_t)s * HID + c] * nrm);
  }
}

__global__ void k_bias_lrelu(float* x, const float* __restrict__ b, int n) {
  int idx = blockIdx.x * 256 + threadIdx.x;
  if (idx < n) {
    float v = x[idx] + b[idx & (HID - 1)];
    x[idx] = (v > 0.f) ? v : 0.01f * v;
  }
}

// ============================================================================
// Feature assembly: [dv(300) | pe(1024) | ec[d](512) | p2d[d](512) |
//                    go[p](512) | d2p[p](512) | zero-pad(20)]
// ============================================================================
__global__ void k_feature(const float* __restrict__ dv, const float* __restrict__ pe,
                          const float* __restrict__ ec, const float* __restrict__ p2d,
                          const float* __restrict__ go, const float* __restrict__ d2p,
                          const int* __restrict__ d_index, const int* __restrict__ p_index,
                          float* __restrict__ feat) {
  int idx = blockIdx.x * 256 + threadIdx.x;
  if (idx >= BATCH * DIMP) return;
  int r = idx / DIMP, c = idx - r * DIMP;
  float v = 0.f;
  if (c < 300)       v = dv[r * 300 + c];
  else if (c < 1324) v = pe[(size_t)r * 1024 + (c - 300)];
  else if (c < 1836) v = ec [(size_t)d_index[r] * HID + (c - 1324)];
  else if (c < 2348) v = p2d[(size_t)d_index[r] * HID + (c - 1836)];
  else if (c < 2860) v = go [(size_t)p_index[r] * HID + (c - 2348)];
  else if (c < DIMF) v = d2p[(size_t)p_index[r] * HID + (c - 2860)];
  feat[idx] = v;
}

// zero-pad Wo1 [3372,1024] -> [3392,1024]
__global__ void k_padW(const float* __restrict__ w, float* __restrict__ wp,
                       int rows, int cols, int prow) {
  int idx = blockIdx.x * 256 + threadIdx.x;
  if (idx < prow * cols) wp[idx] = (idx < rows * cols) ? w[idx] : 0.f;
}

// ============================================================================
// BatchNorm (training mode, biased variance) + ReLU
// ============================================================================
__global__ void k_bnstats(const float* __restrict__ x, int rows, int cols,
                          const float* __restrict__ g, const float* __restrict__ be,
                          float* __restrict__ scale, float* __restrict__ shift) {
  __shared__ float ss[256], sq[256];
  int c = blockIdx.x;
  float s = 0.f, q = 0.f;
  for (int r = threadIdx.x; r < rows; r += 256) {
    float v = x[(size_t)r * cols + c]; s += v; q += v * v;
  }
  ss[threadIdx.x] = s; sq[threadIdx.x] = q;
  __syncthreads();
  for (int o = 128; o > 0; o >>= 1) {
    if (threadIdx.x < o) { ss[threadIdx.x] += ss[threadIdx.x + o];
                           sq[threadIdx.x] += sq[threadIdx.x + o]; }
    __syncthreads();
  }
  if (threadIdx.x == 0) {
    float m   = ss[0] / rows;
    float var = sq[0] / rows - m * m;
    float a   = g[c] * rsqrtf(var + EPS_BN);
    scale[c] = a; shift[c] = be[c] - m * a;
  }
}

__global__ void k_bnrelu(float* x, const float* __restrict__ scale,
                         const float* __restrict__ shift, int n, int colmask) {
  int idx = blockIdx.x * 256 + threadIdx.x;
  if (idx < n) {
    int c = idx & colmask;
    x[idx] = fmaxf(x[idx] * scale[c] + shift[c], 0.f);
  }
}

// final 256 -> 1 projection
__global__ void k_final(const float* __restrict__ h, const float* __restrict__ w,
                        const float* __restrict__ b, float* __restrict__ y) {
  int r = blockIdx.x * 256 + threadIdx.x;
  if (r < BATCH) {
    float s = b[0];
    #pragma unroll 8
    for (int c = 0; c < O2; ++c) s += h[(size_t)r * O2 + c] * w[c];
    y[r] = s;
  }
}

// ============================================================================
extern "C" void kernel_launch(void* const* d_in, const int* in_sizes, int n_in,
                              void* d_out, int out_size, void* d_ws, size_t ws_size,
                              hipStream_t stream) {
  (void)in_sizes; (void)n_in; (void)out_size; (void)ws_size;
  const int*   d_index = (const int*)  d_in[0];
  const int*   p_index = (const int*)  d_in[1];
  const float* d_vecs  = (const float*)d_in[2];
  const float* p_emb   = (const float*)d_in[3];
  const float* d_ecfps = (const float*)d_in[4];
  const int*   d_ei    = (const int*)  d_in[5];
  const float* d_ew    = (const float*)d_in[6];
  const float* p_gos   = (const float*)d_in[7];
  const int*   p_ei    = (const int*)  d_in[8];
  const float* p_ew    = (const float*)d_in[9];
  const float* aff     = (const float*)d_in[10];
  const float* W_e = (const float*)d_in[11]; const float* b_e = (const float*)d_in[12];
  const float* W_g = (const float*)d_in[13]; const float* b_g = (const float*)d_in[14];
  const float* W1  = (const float*)d_in[15]; const float* b1  = (const float*)d_in[16];
  const float* W2  = (const float*)d_in[17]; const float* b2  = (const float*)d_in[18];
  const float* Wo1 = (const float*)d_in[19]; const float* bo1 = (const float*)d_in[20];
  const float* g1  = (const float*)d_in[21]; const float* be1 = (const float*)d_in[22];
  const float* Wo2 = (const float*)d_in[23]; const float* bo2 = (const float*)d_in[24];
  const float* g2  = (const float*)d_in[25]; const float* be2 = (const float*)d_in[26];
  const float* Wo3 = (const float*)d_in[27]; const float* bo3 = (const float*)d_in[28];
  float* out = (float*)d_out;

  // ---- workspace bump allocator (floats) ----
  float* ws = (float*)d_ws;
  size_t off = 0;
  auto alloc = [&](size_t n) { float* p = ws + off; off += (n + 63) & ~(size_t)63; return p; };
  float* he     = alloc((size_t)ND_  * HID);    // h_e, later t1, later d2p
  float* hg     = alloc((size_t)NP__ * HID);    // h_g, later t2
  float* ec     = alloc((size_t)ND_  * HID);    // ecfps (post GCN), later m1
  float* go     = alloc((size_t)NP__ * HID);    // gos   (post GCN), later m2
  float* p2d    = alloc((size_t)ND_  * HID);
  float* feat   = alloc((size_t)BATCH * DIMP);
  float* wo1p   = alloc((size_t)DIMP * O1);
  float* dinv_d = alloc(ND_);
  float* dinv_p = alloc(NP__);
  float* sc1 = alloc(O1); float* sh1 = alloc(O1);
  float* sc2 = alloc(O2); float* sh2 = alloc(O2);
  float* t1 = he; float* t2 = hg; float* d2p = he;
  float* m1 = ec; float* m2 = go;

  const dim3 blk(256);

  // 1) GCN linear transforms: h = X @ W   [8192,1024]x[1024,512]
  wmma_gemm<false,0><<<dim3(HID/128, ND_/128),  blk, 0, stream>>>(d_ecfps, W_e, nullptr, he, 1024, 1024, HID, HID);
  wmma_gemm<false,0><<<dim3(HID/128, NP__/128), blk, 0, stream>>>(p_gos,   W_g, nullptr, hg, 1024, 1024, HID, HID);

  // 2) symmetric-norm degrees (self-loop weight 1)
  k_fill1<<<dim3((ND_ + 255)/256),  blk, 0, stream>>>(dinv_d, ND_);
  k_deg  <<<dim3(ED_/256),          blk, 0, stream>>>(d_ei, d_ew, dinv_d, ED_);
  k_dinv <<<dim3((ND_ + 255)/256),  blk, 0, stream>>>(dinv_d, ND_);
  k_fill1<<<dim3((NP__ + 255)/256), blk, 0, stream>>>(dinv_p, NP__);
  k_deg  <<<dim3(EP__/256),         blk, 0, stream>>>(p_ei, p_ew, dinv_p, EP__);
  k_dinv <<<dim3((NP__ + 255)/256), blk, 0, stream>>>(dinv_p, NP__);

  // 3) scatter-add aggregation + bias + leaky-relu
  k_selfinit  <<<dim3(ND_*HID/256),        blk, 0, stream>>>(he, dinv_d, ec, ND_*HID);
  k_edgeagg   <<<dim3(ED_/EPB, HID/256),   blk, 0, stream>>>(d_ei, d_ew, dinv_d, he, ec, ED_);
  k_bias_lrelu<<<dim3(ND_*HID/256),        blk, 0, stream>>>(ec, b_e, ND_*HID);
  k_selfinit  <<<dim3(NP__*HID/256),       blk, 0, stream>>>(hg, dinv_p, go, NP__*HID);
  k_edgeagg   <<<dim3(EP__/EPB, HID/256),  blk, 0, stream>>>(p_ei, p_ew, dinv_p, hg, go, EP__);
  k_bias_lrelu<<<dim3(NP__*HID/256),       blk, 0, stream>>>(go, b_g, NP__*HID);

  // 4) affinity matmuls (the big ones, K = 8192)
  wmma_gemm<false,0><<<dim3(HID/128, ND_/128),  blk, 0, stream>>>(aff, go, nullptr, t1, NP__, NP__, HID, HID);
  wmma_gemm<true, 0><<<dim3(HID/128, NP__/128), blk, 0, stream>>>(aff, ec, nullptr, t2, ND_,  NP__, HID, HID);

  // 5) p2d = relu(t1 @ W1 + b1), d2p = relu(t2 @ W2 + b2)
  wmma_gemm<false,2><<<dim3(HID/128, ND_/128),  blk, 0, stream>>>(t1, W1, b1, p2d, HID, HID, HID, HID);
  wmma_gemm<false,2><<<dim3(HID/128, NP__/128), blk, 0, stream>>>(t2, W2, b2, d2p, HID, HID, HID, HID);

  // 6) assemble padded feature matrix and pad Wo1
  k_feature<<<dim3((BATCH*DIMP + 255)/256), blk, 0, stream>>>(d_vecs, p_emb, ec, p2d, go, d2p, d_index, p_index, feat);
  k_padW   <<<dim3((DIMP*O1 + 255)/256),    blk, 0, stream>>>(Wo1, wo1p, DIMF, O1, DIMP);

  // 7) MLP layer 1: [4096,3392]x[3392,1024] + bias, then BN + ReLU
  wmma_gemm<false,1><<<dim3(O1/128, BATCH/128), blk, 0, stream>>>(feat, wo1p, bo1, m1, DIMP, DIMP, O1, O1);
  k_bnstats<<<dim3(O1),                blk, 0, stream>>>(m1, BATCH, O1, g1, be1, sc1, sh1);
  k_bnrelu <<<dim3(BATCH*O1/256),      blk, 0, stream>>>(m1, sc1, sh1, BATCH*O1, O1 - 1);

  // 8) MLP layer 2: [4096,1024]x[1024,256] + bias, then BN + ReLU
  wmma_gemm<false,1><<<dim3(O2/128, BATCH/128), blk, 0, stream>>>(m1, Wo2, bo2, m2, O1, O1, O2, O2);
  k_bnstats<<<dim3(O2),                blk, 0, stream>>>(m2, BATCH, O2, g2, be2, sc2, sh2);
  k_bnrelu <<<dim3(BATCH*O2/256),      blk, 0, stream>>>(m2, sc2, sh2, BATCH*O2, O2 - 1);

  // 9) final projection 256 -> 1
  k_final<<<dim3((BATCH + 255)/256), blk, 0, stream>>>(m2, Wo3, bo3, out);
}
